// SkipGram_9586367004719
// MI455X (gfx1250) — matrix operand order repaired and checked by reading, not compiled, not run on hardware
//
#include <hip/hip_runtime.h>
#include <hip/hip_bf16.h>
#include <cmath>

#define VOCAB 100000
#define DIM   128
#define NEG   5
#define NROWS 10240
#define LMAX  8
#define TILE  16
#define NTILES (NROWS / TILE)   // 640

typedef __attribute__((ext_vector_type(16))) _Float16 v16h;
typedef __attribute__((ext_vector_type(8)))  _Float16 v8h;
typedef __attribute__((ext_vector_type(8)))  float    v8f;

// One block (4 waves of 32) per 16-row tile.
// Phase 1: cooperative mean-pooling of 7*16 = 112 vectors. All LMAX=8 row
//          gathers are issued unconditionally (indices are valid in every
//          slot) and masked afterward -> 8 independent global_load_b128 in
//          flight per pooled vector for latency hiding; f32 accumulate,
//          f16 store to LDS.
// Phase 2: wave 0 computes six 16x16 f32 accumulators with
//          v_wmma_f32_16x16x32_f16 over 4 K-chunks; diagonals = dot products.
// Phase 3: per-row loss, per-tile partial sum to workspace.
__global__ __launch_bounds__(128) void skipgram_tile_kernel(
    const float* __restrict__ u_weight,
    const float* __restrict__ v_weight,
    const int* __restrict__ pos_u, const int* __restrict__ len_u,
    const int* __restrict__ pos_v, const int* __restrict__ len_v,
    const int* __restrict__ neg_v, const int* __restrict__ len_neg,
    float* __restrict__ partial)
{
    __shared__ _Float16 sU[TILE * DIM];        // A rows: pooled embed_u (f16)
    __shared__ _Float16 sT[6 * TILE * DIM];    // B cols: t=0 -> v, t=1..5 -> neg_{t-1}
    __shared__ float    sScore[6 * TILE];
    __shared__ float    sLoss[TILE];

    const int tile = blockIdx.x;
    const int n0   = tile * TILE;
    const int tid  = threadIdx.x;
    const int lane = tid & 31;
    const int wave = tid >> 5;

    // ---------------- pooling: 112 vectors, distributed over 4 waves -------
    for (int vid = wave; vid < 7 * TILE; vid += 4) {
        const int g = vid >> 4;            // 0:u, 1:v, 2..6: neg k=g-2
        const int r = vid & (TILE - 1);
        const int n = n0 + r;

        const float* W;
        const int*   idxp;
        int          len;
        _Float16*    dst;
        if (g == 0) {
            W = u_weight; idxp = pos_u + (size_t)n * LMAX; len = len_u[n];
            dst = sU + r * DIM;
        } else if (g == 1) {
            W = v_weight; idxp = pos_v + (size_t)n * LMAX; len = len_v[n];
            dst = sT + r * DIM;
        } else {
            const int m = n * NEG + (g - 2);
            W = v_weight; idxp = neg_v + (size_t)m * LMAX; len = len_neg[m];
            dst = sT + ((g - 1) * TILE + r) * DIM;
        }

        // All 8 indices (uniform, one cacheline), then 8 independent gathers.
        const int4 i0 = *(const int4*)(idxp);
        const int4 i1 = *(const int4*)(idxp + 4);
        int ids[LMAX] = { i0.x, i0.y, i0.z, i0.w, i1.x, i1.y, i1.z, i1.w };

        float4 w[LMAX];
        #pragma unroll
        for (int l = 0; l < LMAX; ++l)
            w[l] = *(const float4*)(W + (size_t)ids[l] * DIM + lane * 4);

        float4 acc = make_float4(0.f, 0.f, 0.f, 0.f);
        #pragma unroll
        for (int l = 0; l < LMAX; ++l) {
            const float m = (l < len) ? 1.0f : 0.0f;
            acc.x += w[l].x * m; acc.y += w[l].y * m;
            acc.z += w[l].z * m; acc.w += w[l].w * m;
        }
        const float inv = 1.0f / (float)len;
        dst[lane * 4 + 0] = (_Float16)(acc.x * inv);
        dst[lane * 4 + 1] = (_Float16)(acc.y * inv);
        dst[lane * 4 + 2] = (_Float16)(acc.z * inv);
        dst[lane * 4 + 3] = (_Float16)(acc.w * inv);
    }
    __syncthreads();

    // ---------------- WMMA: wave 0, full EXEC ------------------------------
    if (tid < 32) {
        const int half_id = lane >> 4;     // K sub-block select per ISA layout
        const int mr      = lane & 15;     // A row / B column held by this lane

        v8f acc[6];
        #pragma unroll
        for (int t = 0; t < 6; ++t) acc[t] = (v8f){};

        #pragma unroll
        for (int kc = 0; kc < 4; ++kc) {
            // A fragment (16-bit A 16x32 layout): lane holds row mr,
            // K groups {half_id*8 .. +7} and {16 + half_id*8 .. +7}
            const _Float16* arow = sU + mr * DIM + kc * 32;
            v8h a0 = *(const v8h*)(arow + half_id * 8);
            v8h a1 = *(const v8h*)(arow + 16 + half_id * 8);
            v16h A;
            #pragma unroll
            for (int i = 0; i < 8; ++i) { A[i] = a0[i]; A[8 + i] = a1[i]; }

            #pragma unroll
            for (int t = 0; t < 6; ++t) {
                // B fragment (32x16): lane holds col mr,
                // K groups {half_id*8 .. +7} and {16 + half_id*8 .. +7}
                const _Float16* bcol =
                    sT + ((t * TILE) + mr) * DIM + kc * 32;
                v8h b0 = *(const v8h*)(bcol + half_id * 8);
                v8h b1 = *(const v8h*)(bcol + 16 + half_id * 8);
                v16h B;
                #pragma unroll
                for (int i = 0; i < 8; ++i) { B[i] = b0[i]; B[8 + i] = b1[i]; }
                acc[t] = __builtin_amdgcn_wmma_f32_16x16x32_f16(
                    false, A, false, B, (short)0, acc[t], false, false);
            }
        }

        // Diagonal of C (f32 16x16): d in 0..7 -> lane d, elem d;
        //                            d in 8..15 -> lane d+16, elem d-8.
        #pragma unroll
        for (int t = 0; t < 6; ++t) {
            #pragma unroll
            for (int gg = 0; gg < 8; ++gg) {
                if (lane == gg)      sScore[t * TILE + gg]     = acc[t][gg];
                if (lane == 24 + gg) sScore[t * TILE + 8 + gg] = acc[t][gg];
            }
        }
    }
    __syncthreads();

    // ---------------- per-row loss -----------------------------------------
    if (tid < TILE) {
        const float score = sScore[tid];
        float se = 0.f;
        #pragma unroll
        for (int k = 1; k < 6; ++k) se += expf(sScore[k * TILE + tid]);
        sLoss[tid] = log1pf(se) - score;
    }
    __syncthreads();
    if (tid == 0) {
        float s = 0.f;
        #pragma unroll
        for (int r = 0; r < TILE; ++r) s += sLoss[r];
        partial[tile] = s;
    }
}

// Deterministic fixed-tree reduction of the 640 tile partials.
__global__ __launch_bounds__(256) void skipgram_reduce_kernel(
    const float* __restrict__ partial, float* __restrict__ out)
{
    __shared__ float buf[256];
    float s = 0.f;
    for (int i = threadIdx.x; i < NTILES; i += 256) s += partial[i];
    buf[threadIdx.x] = s;
    __syncthreads();
    for (int st = 128; st > 0; st >>= 1) {
        if ((int)threadIdx.x < st) buf[threadIdx.x] += buf[threadIdx.x + st];
        __syncthreads();
    }
    if (threadIdx.x == 0) out[0] = buf[0] * (1.0f / 1024.0f);
}

extern "C" void kernel_launch(void* const* d_in, const int* in_sizes, int n_in,
                              void* d_out, int out_size, void* d_ws, size_t ws_size,
                              hipStream_t stream) {
    const float* u_weight = (const float*)d_in[0];
    const float* v_weight = (const float*)d_in[1];
    const int*   pos_u    = (const int*)d_in[2];
    const int*   len_u    = (const int*)d_in[3];
    const int*   pos_v    = (const int*)d_in[4];
    const int*   len_v    = (const int*)d_in[5];
    const int*   neg_v    = (const int*)d_in[6];
    const int*   len_neg  = (const int*)d_in[7];
    float* out     = (float*)d_out;
    float* partial = (float*)d_ws;   // 640 floats, fully rewritten every call

    skipgram_tile_kernel<<<dim3(NTILES), dim3(128), 0, stream>>>(
        u_weight, v_weight, pos_u, len_u, pos_v, len_v, neg_v, len_neg, partial);
    skipgram_reduce_kernel<<<dim3(1), dim3(256), 0, stream>>>(partial, out);
}
